// Network_13056700580610
// MI455X (gfx1250) — matrix operand order repaired and checked by reading, not compiled, not run on hardware
//
#include <hip/hip_runtime.h>

typedef __attribute__((ext_vector_type(16))) _Float16 v16h;
typedef __attribute__((ext_vector_type(8)))  float    v8f;

#define T_STEPS 15
#define C_IN    6
#define C_OUT   64
#define HW      256
#define KS      5
#define KTOT    150          // C_IN*KS*KS
#define KPAD    160          // padded to 5 chunks of 32
#define XSTRIDE 264          // padded-row stride in LDS (260 used)
#define ZBASE   (C_IN*KS*XSTRIDE)       // 7920
#define NLOC    (HW*HW)                 // 65536
#define PLANE   ((size_t)T_STEPS*C_OUT*NLOC)  // 62914560 floats per big output
#define THRESH  15.0f

union AFrag { v16h v; _Float16 h[16]; };
union BFrag { v16h v; int4 q[2]; };

// ---------------------------------------------------------------------------
// Phase 1: implicit-GEMM conv via v_wmma_f32_16x16x32_f16, fused threshold,
// fused per-(t,h,w) channel max/argmax reduction (packed u64 LDS atomicMax).
// grid = T*H blocks, 256 threads (8 waves). Wave w computes w-tiles 2w,2w+1
// (16 output columns each) x all 64 channels (4 N-tiles).
// ---------------------------------------------------------------------------
__global__ __launch_bounds__(256) void conv_kernel(const float* __restrict__ sp,
                                                   const float* __restrict__ w1,
                                                   float* __restrict__ out,
                                                   float* __restrict__ mxmap,
                                                   int*   __restrict__ amxmap) {
    __shared__ _Float16 wlds[C_OUT * KPAD];           // weights, N-major, K padded
    __shared__ _Float16 xlds[ZBASE + 272];            // input rows + zero region
    __shared__ int      kofft[KPAD];                  // K -> LDS element offset
    __shared__ unsigned long long red[HW];            // packed (val, 64-c) per w

    const int tid = threadIdx.x;
    const int t = blockIdx.x / HW;
    const int h = blockIdx.x % HW;

    // --- stage weights: wlds[n*KPAD + k] = f16(w1[n*150 + k]), zeros beyond 150
    for (int i = tid; i < C_OUT * KPAD; i += 256) {
        const int k = i % KPAD, n = i / KPAD;
        float v = (k < KTOT) ? w1[(size_t)n * KTOT + k] : 0.0f;
        wlds[i] = (_Float16)v;
    }
    // --- stage input rows (zero-padded by 2 on each side, 5 rows around h)
    for (int i = tid; i < C_IN * KS * 260; i += 256) {
        const int wp = i % 260;
        const int rr = i / 260;
        const int kh = rr % KS, cin = rr / KS;
        const int hh = h + kh - 2, ww = wp - 2;
        float v = 0.0f;
        if (hh >= 0 && hh < HW && ww >= 0 && ww < HW)
            v = sp[(((size_t)t * C_IN + cin) * HW + hh) * HW + ww];
        xlds[(cin * KS + kh) * XSTRIDE + wp] = (_Float16)v;
    }
    // --- zero region (for padded K >= 150), K->offset table, reduction init
    if (tid < 272) xlds[ZBASE + tid] = (_Float16)0.0f;
    if (tid < KPAD) {
        int off = ZBASE;
        if (tid < KTOT) {
            const int cin = tid / 25, r = tid % 25;
            off = (cin * KS + r / KS) * XSTRIDE + (r % KS);
        }
        kofft[tid] = off;
    }
    red[tid] = 0ULL;
    __syncthreads();

    const int lane = tid & 31;
    const int wid  = tid >> 5;
    const int m    = lane & 15;          // row-within-tile / channel-within-tile
    const int hi   = (lane >> 4) & 1;    // lane half

    v8f acc[2][4] = {};

#pragma unroll
    for (int kc = 0; kc < 5; ++kc) {
        // A-fragment K offsets for this lane (contiguous K order matching B:
        // lanes 0-15 hold K 0-15, lanes 16-31 hold K 16-31)
        const int koffA = kc * 32 + hi * 16;
        int offs[16];
#pragma unroll
        for (int i = 0; i < 8; ++i) {
            offs[i]     = kofft[koffA + i];
            offs[8 + i] = kofft[koffA + 8 + i];
        }
        // B-fragments (contiguous K order, same as A: lanes 0-15 hold K 0-15, lanes
        // 16-31 hold K 16-31, column N = lane&15) -> two aligned b128 loads
        const int kb = kc * 32 + hi * 16;
        BFrag bf[4];
#pragma unroll
        for (int nt = 0; nt < 4; ++nt) {
            const _Float16* bp = &wlds[(nt * 16 + m) * KPAD + kb];
            bf[nt].q[0] = *(const int4*)(bp);
            bf[nt].q[1] = *(const int4*)(bp + 8);
        }
#pragma unroll
        for (int mt = 0; mt < 2; ++mt) {
            const int wcol = (wid * 2 + mt) * 16 + m;   // im2col row = output w
            AFrag af;
#pragma unroll
            for (int i = 0; i < 16; ++i) af.h[i] = xlds[offs[i] + wcol];
#pragma unroll
            for (int nt = 0; nt < 4; ++nt)
                acc[mt][nt] = __builtin_amdgcn_wmma_f32_16x16x32_f16(
                    false, af.v, false, bf[nt].v, (short)0, acc[mt][nt],
                    false, false);
        }
    }

    // --- threshold in place, store pot as b128, feed channel max/argmax
    float* pot = out + PLANE;
#pragma unroll
    for (int mt = 0; mt < 2; ++mt) {
        const int wbase = (wid * 2 + mt) * 16 + hi * 8;
#pragma unroll
        for (int nt = 0; nt < 4; ++nt) {
            const int c = nt * 16 + m;       // N = lane&15 within 16-ch block
            const size_t base = (((size_t)t * C_OUT + c) * HW + h) * HW + wbase;
#pragma unroll
            for (int r = 0; r < 8; ++r) {
                float v = acc[mt][nt][r];
                acc[mt][nt][r] = (v < THRESH) ? 0.0f : v;
            }
            float4 o0 = make_float4(acc[mt][nt][0], acc[mt][nt][1],
                                    acc[mt][nt][2], acc[mt][nt][3]);
            float4 o1 = make_float4(acc[mt][nt][4], acc[mt][nt][5],
                                    acc[mt][nt][6], acc[mt][nt][7]);
            *(float4*)(&pot[base])     = o0;
            *(float4*)(&pot[base + 4]) = o1;
        }
        // per-lane pre-reduction over the 4 channel blocks, then one packed
        // atomicMax per owned w. Key = (f32 bits << 32) | (64 - c): nonneg
        // floats order as uints; equal values pick the smallest channel
        // (matches jnp.argmax first-occurrence tie-break).
#pragma unroll
        for (int r = 0; r < 8; ++r) {
            float bv = -1.0f; int bc = 0;
#pragma unroll
            for (int nt = 0; nt < 4; ++nt) {
                const float v = acc[mt][nt][r];
                if (v > bv) { bv = v; bc = nt * 16 + m; }
            }
            const unsigned long long key =
                ((unsigned long long)__float_as_uint(bv) << 32) |
                (unsigned)(64 - bc);
            atomicMax(&red[wbase + r], key);
        }
    }
    __syncthreads();

    // --- emit channel-max / argmax maps (one entry per w)
    {
        const unsigned long long kk = red[tid];
        const size_t mi = (size_t)t * NLOC + h * HW + tid;
        mxmap[mi]  = __uint_as_float((unsigned)(kk >> 32));
        amxmap[mi] = (64 - (int)(kk & 0xffffffffu)) & 63;
    }
}

// ---------------------------------------------------------------------------
// Phase 2: pointwise inhibition + spikes + compact winner maps.
// Uses conv's max/argmax maps: ~32 reads per pixel instead of ~2000.
// grid = 256 blocks (h), 256 threads (w); each thread owns one pixel column.
// ---------------------------------------------------------------------------
__global__ __launch_bounds__(256) void inhibit_kernel(float* __restrict__ out,
                                                      const float* __restrict__ mxmap,
                                                      const int*   __restrict__ amxmap,
                                                      float* __restrict__ tval,
                                                      float* __restrict__ nspk,
                                                      int*   __restrict__ chmp) {
    const int h = blockIdx.x, w = threadIdx.x;
    const int base = h * HW + w;
    float* pot = out + PLANE;
    float* spk = out;

    // earliest any-channel spike step + last-step gate, from the tiny maps
    int nAny = 0;
    float lastmax = 0.0f;
#pragma unroll
    for (int t = 0; t < T_STEPS; ++t) {
        const float m = mxmap[(size_t)t * NLOC + base];
        if (m > 0.0f) nAny++;
        if (t == T_STEPS - 1) lastmax = m;
    }
    int earliest = T_STEPS - nAny;
    earliest = earliest < 0 ? 0 : (earliest > T_STEPS - 1 ? T_STEPS - 1 : earliest);

    const int cw = amxmap[(size_t)earliest * NLOC + base] & 63;
    const bool gate = lastmax > 0.0f;

    // winner-channel time series (the only pot values that survive)
    float pv[T_STEPS];
    int cnt = 0;
#pragma unroll
    for (int t = 0; t < T_STEPS; ++t) {
        pv[t] = pot[(size_t)(t * C_OUT + cw) * NLOC + base];
        if (gate && pv[t] > 0.0f) cnt++;
    }
    int e2 = T_STEPS - cnt;
    e2 = e2 < 0 ? 0 : (e2 > T_STEPS - 1 ? T_STEPS - 1 : e2);
    const float tv = gate ? pot[(size_t)(e2 * C_OUT + cw) * NLOC + base] : 0.0f;

    tval[base] = tv;
    nspk[base] = (float)cnt;
    chmp[base] = cw;

    // write pass: zeros everywhere except the gated winner channel
    for (int t = 0; t < T_STEPS; ++t)
        for (int c = 0; c < C_OUT; ++c) {
            const size_t idx = (size_t)(t * C_OUT + c) * NLOC + base;
            const float o = (gate && c == cw) ? pv[t] : 0.0f;
            pot[idx] = o;
            spk[idx] = (o > 0.0f) ? 1.0f : 0.0f;
        }
}

// ---------------------------------------------------------------------------
// Phase 3: k-WTA over the compact 65536-entry map. Single 1024-thread block.
// ---------------------------------------------------------------------------
__global__ __launch_bounds__(1024) void winners_kernel(const float* __restrict__ tval,
                                                       const float* __restrict__ nspk,
                                                       const int*   __restrict__ chmp,
                                                       float* __restrict__ out) {
    __shared__ float sval[1024];
    __shared__ int   sidx[1024];
    __shared__ float shv;
    __shared__ int   wc[5], wh[5], ww[5];

    const int tid = threadIdx.x;

    // v = max(spk * values) * T
    float mx = 0.0f;
    for (int i = tid; i < NLOC; i += 1024)
        if (nspk[i] > 0.0f) mx = fmaxf(mx, tval[i]);
    sval[tid] = mx;
    __syncthreads();
    for (int s = 512; s > 0; s >>= 1) {
        if (tid < s) sval[tid] = fmaxf(sval[tid], sval[tid + s]);
        __syncthreads();
    }
    if (tid == 0) shv = sval[0] * (float)T_STEPS;
    __syncthreads();
    const float v = shv;

    float* wout = out + 2 * PLANE;

    for (int k = 0; k < 5; ++k) {
        float best = 0.0f;
        int bflat = 0x7fffffff;
        for (int i = tid; i < NLOC; i += 1024) {
            const float ns = nspk[i];
            const float tot = ns * (tval[i] + v);
            if (tot <= 0.0f) continue;
            const int c = chmp[i];
            const int hh = i >> 8, wcol = i & 255;
            bool sup = false;
            for (int j = 0; j < k; ++j) {
                if (wc[j] < 0) continue;
                if (c == wc[j]) sup = true;                    // channel inhibition
                const int dh = hh - wh[j], dw = wcol - ww[j];
                if (dh >= -3 && dh <= 3 && dw >= -3 && dw <= 3) sup = true; // box
            }
            if (sup) continue;
            const int flat = c * NLOC + i;                    // (C,H,W) flat index
            if (tot > best || (tot == best && flat < bflat)) { best = tot; bflat = flat; }
        }
        sval[tid] = best;
        sidx[tid] = bflat;
        __syncthreads();
        for (int s = 512; s > 0; s >>= 1) {
            if (tid < s) {
                const float ov = sval[tid + s];
                const int   oi = sidx[tid + s];
                if (ov > sval[tid] || (ov == sval[tid] && oi < sidx[tid])) {
                    sval[tid] = ov; sidx[tid] = oi;
                }
            }
            __syncthreads();
        }
        if (tid == 0) {
            if (sval[0] > 0.0f) {
                const int flat = sidx[0];
                const int c = flat >> 16, rem = flat & (NLOC - 1);
                wc[k] = c; wh[k] = rem >> 8; ww[k] = rem & 255;
                wout[k * 3 + 0] = (float)c;
                wout[k * 3 + 1] = (float)(rem >> 8);
                wout[k * 3 + 2] = (float)(rem & 255);
            } else {
                wc[k] = -1; wh[k] = -1; ww[k] = -1;
                wout[k * 3 + 0] = -1.0f;
                wout[k * 3 + 1] = -1.0f;
                wout[k * 3 + 2] = -1.0f;
            }
        }
        __syncthreads();
    }
}

// ---------------------------------------------------------------------------
extern "C" void kernel_launch(void* const* d_in, const int* in_sizes, int n_in,
                              void* d_out, int out_size, void* d_ws, size_t ws_size,
                              hipStream_t stream) {
    const float* sp = (const float*)d_in[0];   // (15,6,256,256) spikes
    const float* w1 = (const float*)d_in[1];   // (64,6,5,5) weights
    float* out = (float*)d_out;                // spk | pot | winners

    // scratch layout (~8.7 MB):
    //   mxmap [15*65536] f32 | amxmap [15*65536] i32 |
    //   tval [65536] f32 | nspk [65536] f32 | chan [65536] i32
    float* mxmap  = (float*)d_ws;
    int*   amxmap = (int*)(mxmap + (size_t)T_STEPS * NLOC);
    float* tval   = (float*)(amxmap + (size_t)T_STEPS * NLOC);
    float* nspk   = tval + NLOC;
    int*   chmp   = (int*)(nspk + NLOC);

    conv_kernel<<<dim3(T_STEPS * HW), dim3(256), 0, stream>>>(sp, w1, out,
                                                              mxmap, amxmap);
    inhibit_kernel<<<dim3(HW), dim3(256), 0, stream>>>(out, mxmap, amxmap,
                                                       tval, nspk, chmp);
    winners_kernel<<<dim3(1), dim3(1024), 0, stream>>>(tval, nspk, chmp, out);
}
